// CausalSelfAttention_13426067767488
// MI455X (gfx1250) — compile-verified, matrix operand order checked by reading, not run on hardware
//
#include <hip/hip_runtime.h>
#include <hip/hip_bf16.h>

// ---------------------------------------------------------------------------
// Types for CDNA5 WMMA (wave32): bf16 A/B fragments (16 x bf16 = 8 VGPRs),
// fp32 C/D accumulators (8 x f32 = 8 VGPRs).
// ---------------------------------------------------------------------------
typedef __attribute__((ext_vector_type(16))) __bf16 bf16x16;
typedef __attribute__((ext_vector_type(8)))  __bf16 bf16x8;
typedef __attribute__((ext_vector_type(8)))  float  f32x8;
typedef __attribute__((ext_vector_type(4)))  int    v4i;

#define WMMA_BF16(a, b, c) \
  __builtin_amdgcn_wmma_f32_16x16x32_bf16(false, (a), false, (b), (short)0, (c), false, false)

// Problem constants (reference: b=2, s=2048, d=1024, 16 heads, hd=64)
#define BB   2
#define SS   2048
#define DD   1024
#define HH   16
#define HD   64
#define MS   (BB * SS)      // 4096 rows for the projection GEMMs

#define ASG __attribute__((address_space(1)))
#define ASL __attribute__((address_space(3)))

// ---------------------------------------------------------------------------
// Async 16-byte global -> LDS copy (ASYNCcnt path, no VGPR bounce) with a
// synchronous fallback if the toolchain lacks the builtin.
// Confirmed signature: (global v4i* src, local v4i* dst, imm offset, imm cpol).
// ---------------------------------------------------------------------------
#if defined(__HIP_DEVICE_COMPILE__) && __has_builtin(__builtin_amdgcn_global_load_async_to_lds_b128)
#define HAVE_ASYNC_LDS 1
#else
#define HAVE_ASYNC_LDS 0
#endif

__device__ __forceinline__ void async_cp16(void* lds, const void* g) {
#if HAVE_ASYNC_LDS
  __builtin_amdgcn_global_load_async_to_lds_b128((ASG v4i*)g, (ASL v4i*)lds, 0, 0);
#else
  *(uint4*)lds = *(const uint4*)g;
#endif
}

__device__ __forceinline__ void async_wait() {
#if HAVE_ASYNC_LDS
#if __has_builtin(__builtin_amdgcn_s_wait_asynccnt)
  __builtin_amdgcn_s_wait_asynccnt(0);
#else
  asm volatile("s_wait_asynccnt 0x0" ::: "memory");
#endif
#endif
}

// ---------------------------------------------------------------------------
// Load one 16x32 bf16 A/B fragment from an LDS tile stored row-major.
// ISA layout (CDNA5 ISA): lanes 0-15 hold row (lane&15), K = {0..7,16..23};
// lanes 16-31 hold the same row, K = {8..15,24..31}. Caller passes
//   p = tile + row*stride + kbase + (lane>>4)*8
// Stride (in elements) must make stride*2 a multiple of 16 bytes.
// ---------------------------------------------------------------------------
__device__ __forceinline__ bf16x16 frag_ld(const __bf16* p) {
  bf16x8 lo = *(const bf16x8*)(p);        // K = k0 .. k0+7
  bf16x8 hi = *(const bf16x8*)(p + 16);   // K = k0+16 .. k0+23
  bf16x16 f;
#pragma unroll
  for (int i = 0; i < 8; ++i) { f[i] = lo[i]; f[i + 8] = hi[i]; }
  return f;
}

// 16-lane (half-wave) XOR reductions via ds_swizzle (group-of-32, and=0x1f).
__device__ __forceinline__ float half_reduce_max(float v) {
  v = fmaxf(v, __int_as_float(__builtin_amdgcn_ds_swizzle(__float_as_int(v), 0x041f)));
  v = fmaxf(v, __int_as_float(__builtin_amdgcn_ds_swizzle(__float_as_int(v), 0x081f)));
  v = fmaxf(v, __int_as_float(__builtin_amdgcn_ds_swizzle(__float_as_int(v), 0x101f)));
  v = fmaxf(v, __int_as_float(__builtin_amdgcn_ds_swizzle(__float_as_int(v), 0x201f)));
  return v;
}
__device__ __forceinline__ float half_reduce_add(float v) {
  v += __int_as_float(__builtin_amdgcn_ds_swizzle(__float_as_int(v), 0x041f));
  v += __int_as_float(__builtin_amdgcn_ds_swizzle(__float_as_int(v), 0x081f));
  v += __int_as_float(__builtin_amdgcn_ds_swizzle(__float_as_int(v), 0x101f));
  v += __int_as_float(__builtin_amdgcn_ds_swizzle(__float_as_int(v), 0x201f));
  return v;
}

// ---------------------------------------------------------------------------
// fp32 -> bf16 elementwise convert
// ---------------------------------------------------------------------------
__global__ void to_bf16_kernel(const float* __restrict__ X, __bf16* __restrict__ Y, int n) {
  int i = blockIdx.x * blockDim.x + threadIdx.x;
  if (i < n) Y[i] = (__bf16)X[i];
}

// ---------------------------------------------------------------------------
// NT GEMM with COMPILE-TIME dims: C[MS,DD] = A[MS,DD] * B[DD,DD]^T.
// A,B bf16 row-major (K contiguous), C fp32 row-major.
// Block = 128x128, K-step 32, 256 threads = 8 waves (2x4 wave grid,
// each wave 64x32 via 4x2 WMMA accumulators).
// LDS tiles are DOUBLE-BUFFERED: the async DMA for tile t+1 is issued before
// the WMMA burst for tile t, so the s_wait_asynccnt lands after the math.
// ---------------------------------------------------------------------------
#define GS 56   // LDS row stride (elements) for 32-wide K tile: 112B = 16B-aligned, conflict-free

__global__ __launch_bounds__(256)
void gemm_nt_bf16(const __bf16* __restrict__ A, const __bf16* __restrict__ B,
                  float* __restrict__ C) {
  constexpr int N = DD, K = DD, KT = K / 32;
  __shared__ __bf16 sA[2][128 * GS];
  __shared__ __bf16 sB[2][128 * GS];

  const int tid  = threadIdx.x;
  const int lane = tid & 31;
  const int wave = tid >> 5;       // 0..7
  const int wm   = wave & 1;       // 2 wave-rows  -> 64 M each
  const int wn   = wave >> 1;      // 4 wave-cols  -> 32 N each
  const int half = lane >> 4;
  const int l16  = lane & 15;
  const int mBase = blockIdx.y * 128;
  const int nBase = blockIdx.x * 128;

  const int lrow = tid >> 2;           // 0..63
  const int lseg = (tid & 3) * 8;      // 8 bf16 (16B) per segment

  auto stage = [&](int k0, int buf) {
#pragma unroll
    for (int p = 0; p < 2; ++p) {
      int r = lrow + p * 64;
      async_cp16(&sA[buf][r * GS + lseg], A + (size_t)(mBase + r) * K + k0 + lseg);
      async_cp16(&sB[buf][r * GS + lseg], B + (size_t)(nBase + r) * K + k0 + lseg);
    }
  };

  f32x8 acc[4][2];
#pragma unroll
  for (int i = 0; i < 4; ++i)
#pragma unroll
    for (int j = 0; j < 2; ++j)
#pragma unroll
      for (int e = 0; e < 8; ++e) acc[i][j][e] = 0.0f;

  // Prologue: stage tile 0.
  stage(0, 0);
  async_wait();
  __syncthreads();

  for (int kt = 0; kt < KT; ++kt) {
    const int cur = kt & 1;
    // Overlap: issue DMA for the next tile into the other buffer.
    if (kt + 1 < KT) stage((kt + 1) * 32, cur ^ 1);

    // Pre-load all fragments of the current tile, then WMMA burst.
    bf16x16 bfrag[2], afrag[4];
#pragma unroll
    for (int j = 0; j < 2; ++j)
      bfrag[j] = frag_ld(&sB[cur][(wn * 32 + j * 16 + l16) * GS + half * 8]);
#pragma unroll
    for (int i = 0; i < 4; ++i)
      afrag[i] = frag_ld(&sA[cur][(wm * 64 + i * 16 + l16) * GS + half * 8]);
#pragma unroll
    for (int i = 0; i < 4; ++i)
#pragma unroll
      for (int j = 0; j < 2; ++j)
        acc[i][j] = WMMA_BF16(afrag[i], bfrag[j], acc[i][j]);

    async_wait();       // next tile's DMA complete (overlapped with math above)
    __syncthreads();    // all waves: next tile visible, current tile consumed
  }

  // Epilogue: C-fragment layout -> row = base + r + 8*half, col = base + (lane&15)
  // N is compile-time, so the 8 stores share one base with immediate offsets.
#pragma unroll
  for (int i = 0; i < 4; ++i)
#pragma unroll
    for (int j = 0; j < 2; ++j) {
      float* cp = C + (size_t)(mBase + wm * 64 + i * 16 + half * 8) * N
                    + (nBase + wn * 32 + j * 16 + l16);
#pragma unroll
      for (int r = 0; r < 8; ++r) cp[r * N] = acc[i][j][r];
    }
}

// ---------------------------------------------------------------------------
// RoPE + repack: read fp32 [b, s, d] (d = h*64 + t), rotate pairs (2i, 2i+1)
// within each head, write bf16 [b, h, s, hd].
// ---------------------------------------------------------------------------
__global__ void rope_pack(const float* __restrict__ X, __bf16* __restrict__ Y) {
  int idx = blockIdx.x * blockDim.x + threadIdx.x;   // ((bb*S + s)*H + h)*32 + i
  if (idx >= BB * SS * HH * (HD / 2)) return;
  int i  = idx & 31;
  int h  = (idx >> 5) & 15;
  int s  = (idx >> 9) & 2047;
  int bb = idx >> 20;

  const float* src = X + ((size_t)(bb * SS + s)) * DD + h * HD + 2 * i;
  float x1 = src[0], x2 = src[1];
  float freq = __powf(10000.0f, -(float)(2 * i) * (1.0f / 64.0f));
  float ang  = (float)s * freq;
  float c = __cosf(ang), sn = __sinf(ang);

  __bf16* dst = Y + (((size_t)(bb * HH + h)) * SS + s) * HD + 2 * i;
  dst[0] = (__bf16)(x1 * c - x2 * sn);
  dst[1] = (__bf16)(x1 * sn + x2 * c);
}

// V repack transposed: fp32 [b, s, d] -> bf16 [b, h, hd, s], so attn@V is NT.
__global__ void v_pack(const float* __restrict__ X, __bf16* __restrict__ Y) {
  int idx = blockIdx.x * blockDim.x + threadIdx.x;   // ((bb*S + s)*H + h)*64 + t
  if (idx >= BB * SS * DD) return;
  int t  = idx & 63;
  int h  = (idx >> 6) & 15;
  int s  = (idx >> 10) & 2047;
  int bb = idx >> 21;
  Y[(((size_t)(bb * HH + h)) * HD + t) * SS + s] =
      (__bf16)X[((size_t)(bb * SS + s)) * DD + h * HD + t];
}

// ---------------------------------------------------------------------------
// Flash attention (causal, online softmax).
//   Q,K: bf16 [b,h,s,64]   V: bf16 [b,h,64,s] (transposed)   O: bf16 [b,s,h,64]
// Block: 128 q-rows, 256 threads = 8 waves; wave w owns q-rows w*16..w*16+15.
// Streams 64-key tiles with DOUBLE-BUFFERED async DMA for K/V.
// S = Q*K^T (NT), ctx += P*V^T (NT, V pre-transposed).
// ---------------------------------------------------------------------------
#define AS 72   // LDS row stride (elements) for 64-wide rows: 144B = 16B-aligned, conflict-free

__global__ __launch_bounds__(256)
void attn_fwd(const __bf16* __restrict__ Q, const __bf16* __restrict__ Kc,
              const __bf16* __restrict__ Vt, __bf16* __restrict__ O) {
  __shared__ __bf16 sQ[128 * AS];
  __shared__ __bf16 sK[2][64 * AS];
  __shared__ __bf16 sV[2][64 * AS];
  __shared__ __bf16 sP[128 * AS];

  const int tid  = threadIdx.x;
  const int lane = tid & 31;
  const int wave = tid >> 5;
  const int half = lane >> 4;
  const int l16  = lane & 15;
  const int qi   = blockIdx.x;           // q tile index (128 rows each)
  const int bh   = blockIdx.y;           // 0..31
  const int q0   = qi * 128;

  const __bf16* Qh = Q  + (size_t)bh * SS * HD;
  const __bf16* Kh = Kc + (size_t)bh * SS * HD;
  const __bf16* Vh = Vt + (size_t)bh * HD * SS;

  auto stageKV = [&](int kbase, int buf) {
    int r   = tid >> 3;
    int seg = (tid & 7) * 8;
#pragma unroll
    for (int p = 0; p < 2; ++p) {
      int rr = r + p * 32;
      async_cp16(&sK[buf][rr * AS + seg], Kh + (size_t)(kbase + rr) * HD + seg);
      async_cp16(&sV[buf][rr * AS + seg], Vh + (size_t)rr * SS + kbase + seg);
    }
  };

  // Prologue: stage Q tile (128x64) and the first K/V tile.
  {
    int r   = tid >> 3;
    int seg = (tid & 7) * 8;
#pragma unroll
    for (int p = 0; p < 4; ++p) {
      int rr = r + p * 32;
      async_cp16(&sQ[rr * AS + seg], Qh + (size_t)(q0 + rr) * HD + seg);
    }
  }
  stageKV(0, 0);
  async_wait();
  __syncthreads();

  f32x8 oacc[4];
#pragma unroll
  for (int dt = 0; dt < 4; ++dt)
#pragma unroll
    for (int e = 0; e < 8; ++e) oacc[dt][e] = 0.0f;

  float m[8], l[8];
#pragma unroll
  for (int r = 0; r < 8; ++r) { m[r] = -__builtin_inff(); l[r] = 0.0f; }

  const float scale = 0.125f;            // 1/sqrt(64)
  const int nKB = qi * 2 + 2;            // 64-key tiles touching this q tile

  for (int kb = 0; kb < nKB; ++kb) {
    const int kbase = kb * 64;
    const int cur = kb & 1;
    // Overlap: issue DMA for the next K/V tile into the other buffer.
    if (kb + 1 < nKB) stageKV(kbase + 64, cur ^ 1);

    // S = Q * K^T : this wave's 16 rows x 64 cols.
    f32x8 sacc[4];
#pragma unroll
    for (int nt = 0; nt < 4; ++nt)
#pragma unroll
      for (int e = 0; e < 8; ++e) sacc[nt][e] = 0.0f;

#pragma unroll
    for (int kk = 0; kk < 2; ++kk) {
      bf16x16 aq = frag_ld(&sQ[(wave * 16 + l16) * AS + kk * 32 + half * 8]);
      bf16x16 bk[4];
#pragma unroll
      for (int nt = 0; nt < 4; ++nt)
        bk[nt] = frag_ld(&sK[cur][(nt * 16 + l16) * AS + kk * 32 + half * 8]);
#pragma unroll
      for (int nt = 0; nt < 4; ++nt)
        sacc[nt] = WMMA_BF16(aq, bk[nt], sacc[nt]);
    }

    // Scale + causal mask + row max.
    float rm[8];
#pragma unroll
    for (int r = 0; r < 8; ++r) rm[r] = -__builtin_inff();
#pragma unroll
    for (int nt = 0; nt < 4; ++nt) {
      int col = kbase + nt * 16 + l16;
#pragma unroll
      for (int r = 0; r < 8; ++r) {
        int rowg = q0 + wave * 16 + half * 8 + r;
        float v = sacc[nt][r] * scale;
        v = (col > rowg) ? -__builtin_inff() : v;
        sacc[nt][r] = v;
        rm[r] = fmaxf(rm[r], v);
      }
    }
#pragma unroll
    for (int r = 0; r < 8; ++r) rm[r] = half_reduce_max(rm[r]);

    float alpha[8];
#pragma unroll
    for (int r = 0; r < 8; ++r) {
      float mn = fmaxf(m[r], rm[r]);
      alpha[r] = __expf(m[r] - mn);      // exp(-inf) = 0 on first tile
      m[r] = mn;
    }

    // P = exp(S - m); wave-private spill to LDS in bf16 for the A fragment.
    float ps[8];
#pragma unroll
    for (int r = 0; r < 8; ++r) ps[r] = 0.0f;
#pragma unroll
    for (int nt = 0; nt < 4; ++nt)
#pragma unroll
      for (int r = 0; r < 8; ++r) {
        float p = __expf(sacc[nt][r] - m[r]);
        ps[r] += p;
        sP[(wave * 16 + half * 8 + r) * AS + nt * 16 + l16] = (__bf16)p;
      }
#pragma unroll
    for (int r = 0; r < 8; ++r) {
      float rs = half_reduce_add(ps[r]);
      l[r] = l[r] * alpha[r] + rs;
    }
#pragma unroll
    for (int dt = 0; dt < 4; ++dt)
#pragma unroll
      for (int r = 0; r < 8; ++r) oacc[dt][r] *= alpha[r];

    // ctx += P * V^T (same-wave DS RAW on sP; hardware DScnt keeps order).
#pragma unroll
    for (int kk = 0; kk < 2; ++kk) {
      bf16x16 ap = frag_ld(&sP[(wave * 16 + l16) * AS + kk * 32 + half * 8]);
      bf16x16 bv[4];
#pragma unroll
      for (int dt = 0; dt < 4; ++dt)
        bv[dt] = frag_ld(&sV[cur][(dt * 16 + l16) * AS + kk * 32 + half * 8]);
#pragma unroll
      for (int dt = 0; dt < 4; ++dt)
        oacc[dt] = WMMA_BF16(ap, bv[dt], oacc[dt]);
    }

    async_wait();       // next K/V tile DMA complete (overlapped with math)
    __syncthreads();    // all waves: next tile visible, current tile consumed
  }

  // Epilogue: normalize and write bf16 ctx in [b, s, h, 64] layout.
  int bb = bh >> 4, hh = bh & 15;
#pragma unroll
  for (int dt = 0; dt < 4; ++dt) {
    int t = dt * 16 + l16;
#pragma unroll
    for (int r = 0; r < 8; ++r) {
      int seq = q0 + wave * 16 + half * 8 + r;
      float inv = 1.0f / l[r];
      O[(((size_t)bb * SS + seq) * HH + hh) * HD + t] = (__bf16)(oacc[dt][r] * inv);
    }
  }
}

// ---------------------------------------------------------------------------
// Host-side orchestration (graph-capture safe: only kernel launches on stream)
// ---------------------------------------------------------------------------
extern "C" void kernel_launch(void* const* d_in, const int* in_sizes, int n_in,
                              void* d_out, int out_size, void* d_ws, size_t ws_size,
                              hipStream_t stream) {
  (void)in_sizes; (void)n_in; (void)out_size; (void)ws_size;

  const float* x  = (const float*)d_in[0];
  const float* Wq = (const float*)d_in[1];
  const float* Wk = (const float*)d_in[2];
  const float* Wv = (const float*)d_in[3];
  const float* Wo = (const float*)d_in[4];
  float* out = (float*)d_out;

  char* ws = (char*)d_ws;
  size_t off = 0;
  auto alloc = [&](size_t bytes) -> char* {
    char* p = ws + off;
    off += (bytes + 255) & ~(size_t)255;
    return p;
  };

  __bf16* xb   = (__bf16*)alloc((size_t)MS * DD * 2);   // 8 MiB
  __bf16* wqb  = (__bf16*)alloc((size_t)DD * DD * 2);   // 2 MiB each
  __bf16* wkb  = (__bf16*)alloc((size_t)DD * DD * 2);
  __bf16* wvb  = (__bf16*)alloc((size_t)DD * DD * 2);
  __bf16* wob  = (__bf16*)alloc((size_t)DD * DD * 2);
  float*  q32  = (float*) alloc((size_t)MS * DD * 4);   // 16 MiB each
  float*  k32  = (float*) alloc((size_t)MS * DD * 4);
  float*  v32  = (float*) alloc((size_t)MS * DD * 4);
  __bf16* qb   = (__bf16*)alloc((size_t)MS * DD * 2);   // [b,h,s,64]
  __bf16* kbuf = (__bf16*)alloc((size_t)MS * DD * 2);   // [b,h,s,64]
  __bf16* vtb  = (__bf16*)alloc((size_t)MS * DD * 2);   // [b,h,64,s]
  __bf16* ctxb = (__bf16*)alloc((size_t)MS * DD * 2);   // [b,s,h,64]

  const int nx = MS * DD, nw = DD * DD;
  to_bf16_kernel<<<(nx + 255) / 256, 256, 0, stream>>>(x,  xb,  nx);
  to_bf16_kernel<<<(nw + 255) / 256, 256, 0, stream>>>(Wq, wqb, nw);
  to_bf16_kernel<<<(nw + 255) / 256, 256, 0, stream>>>(Wk, wkb, nw);
  to_bf16_kernel<<<(nw + 255) / 256, 256, 0, stream>>>(Wv, wvb, nw);
  to_bf16_kernel<<<(nw + 255) / 256, 256, 0, stream>>>(Wo, wob, nw);

  dim3 gg(DD / 128, MS / 128);                       // (8, 32)
  gemm_nt_bf16<<<gg, 256, 0, stream>>>(xb, wqb, q32);
  gemm_nt_bf16<<<gg, 256, 0, stream>>>(xb, wkb, k32);
  gemm_nt_bf16<<<gg, 256, 0, stream>>>(xb, wvb, v32);

  const int nrope = BB * SS * HH * (HD / 2);
  rope_pack<<<(nrope + 255) / 256, 256, 0, stream>>>(q32, qb);
  rope_pack<<<(nrope + 255) / 256, 256, 0, stream>>>(k32, kbuf);
  v_pack<<<(nx + 255) / 256, 256, 0, stream>>>(v32, vtb);

  attn_fwd<<<dim3(SS / 128, BB * HH), 256, 0, stream>>>(qb, kbuf, vtb, ctxb);

  gemm_nt_bf16<<<gg, 256, 0, stream>>>(ctxb, wob, out);
}
